// SequencePrediction_14078902796769
// MI455X (gfx1250) — matrix operand order repaired and checked by reading, not compiled, not run on hardware
//
#include <hip/hip_runtime.h>
#include <hip/hip_bf16.h>

#define H    51
#define G4   204              // 4*H gates
#define GP   208              // padded gate dim = 13 * 16
#define NT   13               // gate N-tiles of 16
#define KP1  64               // padded K for cell1: h1(51) | x | 1 | zeros
#define KP2  128              // padded K for cell2: h1|1|0 (64) | h2|0 (64)
#define BATCH 4096
#define TLEN  512
#define FUT   16
#define TOUT  (TLEN + FUT)
#define ROWS  16              // batch rows per block (one wave32)

typedef __attribute__((ext_vector_type(16))) _Float16 v16h;
typedef __attribute__((ext_vector_type(8)))  _Float16 v8h;
typedef __attribute__((ext_vector_type(8)))  float    v8f;

// Fast saturating activations on the serial path: v_exp_f32 + v_rcp_f32 only.
// sig(x)  = rcp(1 + e^-x)            : exact 0/1 limits (rcp(inf)=0)
// tanh(x) = 1 - 2*rcp(1 + e^{2x})    : exact -1/+1 limits, no inf/inf NaN
__device__ __forceinline__ float fast_rcp(float x) { return __builtin_amdgcn_rcpf(x); }
__device__ __forceinline__ float sigf(float x)     { return fast_rcp(1.0f + __expf(-x)); }
__device__ __forceinline__ float tanhfast(float x) { return 1.0f - 2.0f * fast_rcp(1.0f + __expf(2.0f * x)); }

__device__ __forceinline__ v16h cat8(v8h lo, v8h hi) {
  v16h r;
#pragma unroll
  for (int i = 0; i < 8; ++i) { r[i] = lo[i]; r[i + 8] = hi[i]; }
  return r;
}

// A fragment (16x32 f16, MxK) per ISA 7.12.2:
//  lane<16 : M=lane,    K = 32*kc + {0..7, 16..23}
//  lane>=16: M=lane-16, K = 32*kc + {8..15, 24..31}
__device__ __forceinline__ v16h load_a(const _Float16* buf, int stride, int kc, int lane) {
  int base = (lane & 15) * stride + kc * 32 + ((lane & 16) ? 8 : 0);
  v8h lo = *(const v8h*)(buf + base);
  v8h hi = *(const v8h*)(buf + base + 16);
  return cat8(lo, hi);
}

// B fragment (32x16 f16, KxN); weights in LDS as [n][k], k contiguous:
//  lane<16 : N=lane,    K = 32*kc + {0..7, 16..23}
//  lane>=16: N=lane-16, K = 32*kc + {8..15, 24..31}
__device__ __forceinline__ v16h load_b(const _Float16* w, int stride, int ntile, int kc, int lane) {
  int base = (ntile * 16 + (lane & 15)) * stride + kc * 32 + ((lane & 16) ? 8 : 0);
  v8h lo = *(const v8h*)(w + base);
  v8h hi = *(const v8h*)(w + base + 16);
  return cat8(lo, hi);
}

__global__ __launch_bounds__(32)
void lstm_seq_kernel(const float* __restrict__ input,
                     const float* __restrict__ Wih1, const float* __restrict__ Whh1,
                     const float* __restrict__ bih1, const float* __restrict__ bhh1,
                     const float* __restrict__ Wih2, const float* __restrict__ Whh2,
                     const float* __restrict__ bih2, const float* __restrict__ bhh2,
                     const float* __restrict__ Wlin, const float* __restrict__ blin,
                     float* __restrict__ out)
{
  __shared__ __align__(16) _Float16 sB1[GP * KP1];   // fused cell1 weights [n][k]
  __shared__ __align__(16) _Float16 sB2[GP * KP2];   // fused cell2 weights [n][k]
  __shared__ __align__(32) float    sWlin[64];       // W_lin padded with zeros
  __shared__ __align__(16) float    sGates[ROWS * GP];
  __shared__ __align__(16) float    sC1[ROWS * H];
  __shared__ __align__(16) float    sC2[ROWS * H];
  __shared__ __align__(16) _Float16 sH1[ROWS * KP1]; // A operand cell1: h1 | x | 1 | 0
  __shared__ __align__(16) _Float16 sHC[ROWS * KP2]; // A operand cell2: h1 | 1 | 0 | h2 | 0

  const int lane = threadIdx.x;
  const int row0 = blockIdx.x * ROWS;

  // ---- one-time: stage fused f16 weight images into LDS, zero state ----
  for (int i = lane; i < GP * KP1; i += 32) {
    int n = i / KP1, k = i % KP1;
    float v = 0.f;
    if (n < G4) {
      if (k < H)           v = Whh1[n * H + k];
      else if (k == H)     v = Wih1[n];                 // scalar-input column
      else if (k == H + 1) v = bih1[n] + bhh1[n];       // bias column (A holds 1.0)
    }
    sB1[i] = (_Float16)v;
  }
  for (int i = lane; i < GP * KP2; i += 32) {
    int n = i / KP2, k = i % KP2;
    float v = 0.f;
    if (n < G4) {
      if (k < H)                      v = Wih2[n * H + k];          // vs h1
      else if (k == H)                v = bih2[n] + bhh2[n];        // bias column
      else if (k >= 64 && k < 64 + H) v = Whh2[n * H + (k - 64)];   // vs h2
    }
    sB2[i] = (_Float16)v;
  }
  for (int i = lane; i < 64; i += 32)       sWlin[i] = (i < H) ? Wlin[i] : 0.f;
  for (int i = lane; i < ROWS * H; i += 32) { sC1[i] = 0.f; sC2[i] = 0.f; }
  for (int i = lane; i < ROWS * KP1; i += 32) sH1[i] = (_Float16)((i % KP1 == H + 1) ? 1.f : 0.f);
  for (int i = lane; i < ROWS * KP2; i += 32) sHC[i] = (_Float16)((i % KP2 == H) ? 1.f : 0.f);
  __syncthreads();

  const float bl = blin[0];
  float xout = 0.f;   // lanes 0..15: feedback output for autoregressive tail

  for (int t = 0; t < TOUT; ++t) {
    // scalar input for this step into the augmented-K column
    if (lane < ROWS) {
      float x = (t < TLEN) ? input[(row0 + lane) * TLEN + t] : xout;
      sH1[lane * KP1 + H] = (_Float16)x;
      if (t + 16 < TLEN) __builtin_prefetch(&input[(row0 + lane) * TLEN + t + 16], 0, 1);
    }
    __syncthreads();

    // ---- cell 1 gates: [16x64] x [64x208] via 13 x 2 WMMAs ----
    {
      v16h a0 = load_a(sH1, KP1, 0, lane);
      v16h a1 = load_a(sH1, KP1, 1, lane);
#pragma unroll
      for (int nt = 0; nt < NT; ++nt) {
        v8f acc = {};
        acc = __builtin_amdgcn_wmma_f32_16x16x32_f16(false, a0, false, load_b(sB1, KP1, nt, 0, lane), (short)0, acc, false, false);
        acc = __builtin_amdgcn_wmma_f32_16x16x32_f16(false, a1, false, load_b(sB1, KP1, nt, 1, lane), (short)0, acc, false, false);
        int n  = nt * 16 + (lane & 15);
        int mb = (lane & 16) ? 8 : 0;
#pragma unroll
        for (int v = 0; v < 8; ++v) sGates[(mb + v) * GP + n] = acc[v];
      }
    }
    __syncthreads();

    // ---- cell 1 pointwise (i,f,g,o recombination across lanes via LDS) ----
    for (int e = lane; e < ROWS * H; e += 32) {
      int m = e / H, k = e - m * H;
      float gi = sGates[m * GP + k];
      float gf = sGates[m * GP + H + k];
      float gg = sGates[m * GP + 2 * H + k];
      float go = sGates[m * GP + 3 * H + k];
      float c  = sigf(gf) * sC1[e] + sigf(gi) * tanhfast(gg);
      sC1[e] = c;
      _Float16 h = (_Float16)(sigf(go) * tanhfast(c));
      sH1[m * KP1 + k] = h;   // next step's cell1 A
      sHC[m * KP2 + k] = h;   // this step's cell2 A
    }
    __syncthreads();

    // ---- cell 2 gates: [16x128] x [128x208] via 13 x 4 WMMAs ----
    {
      v16h a0 = load_a(sHC, KP2, 0, lane);
      v16h a1 = load_a(sHC, KP2, 1, lane);
      v16h a2 = load_a(sHC, KP2, 2, lane);
      v16h a3 = load_a(sHC, KP2, 3, lane);
#pragma unroll
      for (int nt = 0; nt < NT; ++nt) {
        v8f acc = {};
        acc = __builtin_amdgcn_wmma_f32_16x16x32_f16(false, a0, false, load_b(sB2, KP2, nt, 0, lane), (short)0, acc, false, false);
        acc = __builtin_amdgcn_wmma_f32_16x16x32_f16(false, a1, false, load_b(sB2, KP2, nt, 1, lane), (short)0, acc, false, false);
        acc = __builtin_amdgcn_wmma_f32_16x16x32_f16(false, a2, false, load_b(sB2, KP2, nt, 2, lane), (short)0, acc, false, false);
        acc = __builtin_amdgcn_wmma_f32_16x16x32_f16(false, a3, false, load_b(sB2, KP2, nt, 3, lane), (short)0, acc, false, false);
        int n  = nt * 16 + (lane & 15);
        int mb = (lane & 16) ? 8 : 0;
#pragma unroll
        for (int v = 0; v < 8; ++v) sGates[(mb + v) * GP + n] = acc[v];
      }
    }
    __syncthreads();

    // ---- cell 2 pointwise ----
    for (int e = lane; e < ROWS * H; e += 32) {
      int m = e / H, k = e - m * H;
      float gi = sGates[m * GP + k];
      float gf = sGates[m * GP + H + k];
      float gg = sGates[m * GP + 2 * H + k];
      float go = sGates[m * GP + 3 * H + k];
      float c  = sigf(gf) * sC2[e] + sigf(gi) * tanhfast(gg);
      sC2[e] = c;
      sHC[m * KP2 + 64 + k] = (_Float16)(sigf(go) * tanhfast(c));  // h2
    }
    __syncthreads();

    // ---- linear head: out = h2 . Wlin + b, vectorized (cols 51..55 are 0 both sides)
    if (lane < ROWS) {
      const v8h* hv = (const v8h*)(sHC + lane * KP2 + 64);
      const v8f* wv = (const v8f*)sWlin;
      float s = bl;
#pragma unroll
      for (int c8 = 0; c8 < 7; ++c8) {
        v8h h8 = hv[c8];
        v8f w8 = wv[c8];
#pragma unroll
        for (int j = 0; j < 8; ++j) s += (float)h8[j] * w8[j];
      }
      xout = s;
      out[(row0 + lane) * TOUT + t] = s;
    }
    // no trailing barrier needed: next step only writes sH1[.,H] before the next
    // __syncthreads(), which does not alias the sHC/sWlin reads above
  }
}

extern "C" void kernel_launch(void* const* d_in, const int* in_sizes, int n_in,
                              void* d_out, int out_size, void* d_ws, size_t ws_size,
                              hipStream_t stream) {
  const float* input = (const float*)d_in[0];
  const float* Wih1  = (const float*)d_in[1];
  const float* Whh1  = (const float*)d_in[2];
  const float* bih1  = (const float*)d_in[3];
  const float* bhh1  = (const float*)d_in[4];
  const float* Wih2  = (const float*)d_in[5];
  const float* Whh2  = (const float*)d_in[6];
  const float* bih2  = (const float*)d_in[7];
  const float* bhh2  = (const float*)d_in[8];
  const float* Wlin  = (const float*)d_in[9];
  const float* blin  = (const float*)d_in[10];
  // d_in[11] = future (16), hardcoded as FUT

  dim3 grid(BATCH / ROWS);
  dim3 block(32);
  hipLaunchKernelGGL(lstm_seq_kernel, grid, block, 0, stream,
                     input, Wih1, Whh1, bih1, bhh1,
                     Wih2, Whh2, bih2, bhh2, Wlin, blin,
                     (float*)d_out);
}
